// Guide_46548855554215
// MI455X (gfx1250) — compile-verified, matrix operand order checked by reading, not run on hardware
//
#include <hip/hip_runtime.h>
#include <hip/hip_bf16.h>

#define BATCH 4096
#define NGATE 1024
#define KSTEP 288   // 256 h + 16 samp + 16 zero-pad
#define KOBS  416   // 400 obs_emb + 16 zero-pad
#define OUTC  15

typedef _Float16 f16;
typedef f16  v16h __attribute__((ext_vector_type(16)));
typedef f16  v8h  __attribute__((ext_vector_type(8)));
typedef float v8f __attribute__((ext_vector_type(8)));

// ---------------------------------------------------------------------------
// Weight pre-swizzle into WMMA B-matrix lane layout (16x16x32 f16).
// B element (k,n) within a 32x16 tile lives at: lane = (n%16) + 16*((k>>3)&1),
// v16h element e with k-offset = (lane>=16?8:0) + (e&7) + (e>=8?16:0).
// Packed half index: (((kt*64 + nt)*32 + lane)*16 + e)
// ---------------------------------------------------------------------------
__global__ __launch_bounds__(256)
void k_pack_wstep(const float* __restrict__ W_hh, const float* __restrict__ W_ih,
                  f16* __restrict__ Bp) {
  int kt = blockIdx.x, nt = blockIdx.y;                 // grid (9, 64)
  for (int idx = threadIdx.x; idx < 512; idx += 256) {
    int lane = idx >> 4, e = idx & 15;
    int n = nt * 16 + (lane & 15);
    int k = kt * 32 + ((lane >= 16) ? 8 : 0) + (e & 7) + ((e >= 8) ? 16 : 0);
    float w = 0.f;
    if (k < 256)      w = W_hh[(size_t)n * 256 + k];
    else if (k < 272) w = W_ih[(size_t)n * 432 + 400 + (k - 256)];
    Bp[((((size_t)kt * 64 + nt) * 32 + lane) << 4) + e] = (f16)w;
  }
}

__global__ __launch_bounds__(256)
void k_pack_wobs(const float* __restrict__ W_ih, f16* __restrict__ Bp) {
  int kt = blockIdx.x, nt = blockIdx.y;                 // grid (13, 64)
  for (int idx = threadIdx.x; idx < 512; idx += 256) {
    int lane = idx >> 4, e = idx & 15;
    int n = nt * 16 + (lane & 15);
    int k = kt * 32 + ((lane >= 16) ? 8 : 0) + (e & 7) + ((e >= 8) ? 16 : 0);
    float w = (k < 400) ? W_ih[(size_t)n * 432 + k] : 0.f;
    Bp[((((size_t)kt * 64 + nt) * 32 + lane) << 4) + e] = (f16)w;
  }
}

// Per-step constant gate bias: b_ih + b_hh + addr_emb[aid]·W_ih[:,416:432]^T
__global__ __launch_bounds__(256)
void k_addrconst(const float* __restrict__ b_ih, const float* __restrict__ b_hh,
                 const float* __restrict__ W_ih, const float* __restrict__ addr_emb,
                 float* __restrict__ out /*[7][1024]*/) {
  int idx = blockIdx.x * 256 + threadIdx.x;
  if (idx >= 7 * 1024) return;
  int s = idx >> 10, j = idx & 1023;
  const int aid[7] = {0, 1, 4, 2, 3, 5, 6};
  float v = b_ih[j] + b_hh[j];
  const float* ae = addr_emb + aid[s] * 16;
  const float* wr = W_ih + (size_t)j * 432 + 416;
  #pragma unroll
  for (int k = 0; k < 16; ++k) v += ae[k] * wr[k];
  out[idx] = v;
}

// ---------------------------------------------------------------------------
// Fused conv1(s2,p1)+relu -> conv2(s2,p1)+relu -> 3x3/3 avgpool -> obs_emb f16
// One block (256 thr) per image. ~46KB LDS.
// ---------------------------------------------------------------------------
__global__ __launch_bounds__(256)
void k_conv(const float* __restrict__ obs, const float* __restrict__ w1,
            const float* __restrict__ b1, const float* __restrict__ w2,
            const float* __restrict__ b2, f16* __restrict__ obs16) {
  __shared__ float s_obs[64 * 64];        // 16KB
  __shared__ float s_c1[3][32][32];       // 12KB rolling conv1 rows
  __shared__ float s_c2[16][16][16];      // 16KB conv2 plane [ch][y][x]
  __shared__ float s_w1[288];
  __shared__ float s_b1[32];
  __shared__ float s_b2[16];
  int t = threadIdx.x, b = blockIdx.x;
  const float* ob = obs + (size_t)b * 4096;
  for (int i = t; i < 4096; i += 256) s_obs[i] = ob[i];
  for (int i = t; i < 288; i += 256) s_w1[i] = w1[i];
  if (t < 32) s_b1[t] = b1[t];
  if (t < 16) s_b2[t] = b2[t];
  __syncthreads();
  for (int y2 = 0; y2 < 16; ++y2) {
    // conv1 rows 2*y2-1 .. 2*y2+1 (out-of-range rows = zero padding)
    for (int i = t; i < 3 * 32 * 32; i += 256) {
      int slot = i >> 10, ch = (i >> 5) & 31, x1 = i & 31;
      int r1 = 2 * y2 - 1 + slot;
      float acc = 0.f;
      if (r1 >= 0 && r1 < 32) {
        acc = s_b1[ch];
        #pragma unroll
        for (int ky = 0; ky < 3; ++ky) {
          int iy = 2 * r1 - 1 + ky;
          if (iy < 0 || iy >= 64) continue;
          #pragma unroll
          for (int kx = 0; kx < 3; ++kx) {
            int ix = 2 * x1 - 1 + kx;
            if (ix < 0 || ix >= 64) continue;
            acc += s_w1[ch * 9 + ky * 3 + kx] * s_obs[iy * 64 + ix];
          }
        }
        acc = fmaxf(acc, 0.f);
      }
      s_c1[slot][ch][x1] = acc;
    }
    __syncthreads();
    {
      int oc = t >> 4, x2 = t & 15;
      float acc = s_b2[oc];
      for (int ic = 0; ic < 32; ++ic) {
        #pragma unroll
        for (int ky = 0; ky < 3; ++ky) {
          #pragma unroll
          for (int kx = 0; kx < 3; ++kx) {
            int ix = 2 * x2 - 1 + kx;
            float v = (ix >= 0 && ix < 32) ? s_c1[ky][ic][ix] : 0.f;
            acc += w2[(oc * 32 + ic) * 9 + ky * 3 + kx] * v;
          }
        }
      }
      s_c2[oc][y2][x2] = fmaxf(acc, 0.f);
    }
    __syncthreads();
  }
  f16* o16 = obs16 + (size_t)b * KOBS;
  for (int i = t; i < KOBS; i += 256) {
    float v = 0.f;
    if (i < 400) {
      int ch = i / 25, rem = i % 25, py = rem / 5, px = rem % 5;
      float s = 0.f;
      #pragma unroll
      for (int dy = 0; dy < 3; ++dy)
        #pragma unroll
        for (int dx = 0; dx < 3; ++dx) s += s_c2[ch][3 * py + dy][3 * px + dx];
      v = s * (1.f / 9.f);
    }
    o16[i] = (f16)v;
  }
}

// ---------------------------------------------------------------------------
// WMMA GEMM: C[M][1024] = A[M][Ka](f16 row-major) x Bpack (pre-swizzled).
// Block = 256 thr = 8 waves; wave -> one 16x16 D tile; grid (M/16, 1024/128).
// ---------------------------------------------------------------------------
template <int KT>
__global__ __launch_bounds__(256)
void k_gemm_wmma(const f16* __restrict__ A, int Ka, const f16* __restrict__ Bp,
                 float* __restrict__ C) {
  const int N = NGATE, Ntiles = N >> 4;
  int lane = threadIdx.x & 31, wave = threadIdx.x >> 5;
  int mt = blockIdx.x, nt = (blockIdx.y << 3) + wave;
  int m = (mt << 4) + (lane & 15);
  const f16* Arow = A + (size_t)m * Ka + ((lane >= 16) ? 8 : 0);
  const f16* Bl = Bp + (((size_t)nt * 32 + lane) << 4);
  const size_t bstride = (size_t)Ntiles * 32 * 16;   // halfs per k-tile
  v8f acc = {};
  #pragma unroll
  for (int kt = 0; kt < KT; ++kt) {
    v8h alo = *reinterpret_cast<const v8h*>(Arow + kt * 32);
    v8h ahi = *reinterpret_cast<const v8h*>(Arow + kt * 32 + 16);
    v16h a = __builtin_shufflevector(alo, ahi, 0, 1, 2, 3, 4, 5, 6, 7, 8, 9,
                                     10, 11, 12, 13, 14, 15);
    v16h bfrag = *reinterpret_cast<const v16h*>(Bl + (size_t)kt * bstride);
    acc = __builtin_amdgcn_wmma_f32_16x16x32_f16(false, a, false, bfrag,
                                                 (short)0, acc, false, false);
  }
  int n0 = (nt << 4) + (lane & 15);
  int mrow = (mt << 4) + ((lane >= 16) ? 8 : 0);
  #pragma unroll
  for (int r = 0; r < 8; ++r)
    C[(size_t)(mrow + r) * N + n0] = acc[r];
}

// xin[b][288] = [h(256 f16) | samp(16) | 0(16)]
__global__ __launch_bounds__(256)
void k_build_xin(f16* __restrict__ xin, const f16* __restrict__ h16, int mode,
                 const float* __restrict__ table, const int* __restrict__ sel,
                 const float* __restrict__ direct) {
  int idx = blockIdx.x * 256 + threadIdx.x;
  if (idx >= BATCH * KSTEP) return;
  int b = idx / KSTEP, col = idx % KSTEP;
  f16 v = (f16)0.f;
  if (col < 256) {
    if (h16) v = h16[(size_t)b * 256 + col];
  } else if (col < 272) {
    int j = col - 256;
    if (mode == 1)      v = (f16)table[sel[b] * 16 + j];
    else if (mode == 2) v = (f16)direct[(size_t)b * 16 + j];
  }
  xin[idx] = v;
}

__global__ __launch_bounds__(256)
void k_lstm(const float* __restrict__ gates, const float* __restrict__ base,
            const float* __restrict__ addrv, const float* __restrict__ c_prev,
            float* __restrict__ c_out, f16* __restrict__ h16) {
  int idx = blockIdx.x * 256 + threadIdx.x;
  if (idx >= BATCH * 256) return;
  int b = idx >> 8, u = idx & 255;
  size_t g0 = (size_t)b * 1024;
  float gi = gates[g0 + u]       + base[g0 + u]       + addrv[u];
  float gf = gates[g0 + 256 + u] + base[g0 + 256 + u] + addrv[256 + u];
  float gg = gates[g0 + 512 + u] + base[g0 + 512 + u] + addrv[512 + u];
  float go = gates[g0 + 768 + u] + base[g0 + 768 + u] + addrv[768 + u];
  float cp = c_prev ? c_prev[idx] : 0.f;
  float si = 1.f / (1.f + expf(-gi));
  float sf = 1.f / (1.f + expf(-gf));
  float so = 1.f / (1.f + expf(-go));
  float c2 = sf * cp + si * tanhf(gg);
  c_out[idx] = c2;
  h16[idx] = (f16)(so * tanhf(c2));
}

// out[b][out_col+n] = h·w[n] + bias[n]  (small N heads)
__global__ __launch_bounds__(256)
void k_head(const f16* __restrict__ h16, const float* __restrict__ w,
            const float* __restrict__ bias, int N, float* __restrict__ out,
            int out_col, int out_stride) {
  int idx = blockIdx.x * 256 + threadIdx.x;
  if (idx >= BATCH * N) return;
  int b = idx / N, n = idx % N;
  const f16* hr = h16 + (size_t)b * 256;
  const float* wr = w + (size_t)n * 256;
  float acc = bias[n];
  for (int k = 0; k < 256; ++k) acc += (float)hr[k] * wr[k];
  out[(size_t)b * out_stride + out_col + n] = acc;
}

__global__ __launch_bounds__(256)
void k_rp_post(const float* __restrict__ raw, const float* __restrict__ eps,
               float* __restrict__ dout, int col, float* __restrict__ save2) {
  int idx = blockIdx.x * 256 + threadIdx.x;
  if (idx >= BATCH * 2) return;
  int b = idx >> 1, j = idx & 1;
  float v = raw[b * 4 + j] + expf(raw[b * 4 + 2 + j]) * eps[b * 2 + j];
  dout[(size_t)b * OUTC + col + j] = v;
  if (save2) save2[b * 2 + j] = v;
}

__global__ __launch_bounds__(128)
void k_mlp(const float* __restrict__ rp0, const float* __restrict__ w1,
           const float* __restrict__ b1, const float* __restrict__ w2,
           const float* __restrict__ b2, const float* __restrict__ w3,
           const float* __restrict__ b3, float* __restrict__ emb) {
  __shared__ float z1[100];
  __shared__ float z2[100];
  int b = blockIdx.x, t = threadIdx.x;
  float x0 = rp0[b * 2], x1 = rp0[b * 2 + 1];
  if (t < 100) z1[t] = tanhf(b1[t] + w1[t * 2] * x0 + w1[t * 2 + 1] * x1);
  __syncthreads();
  if (t < 100) {
    float a = b2[t];
    for (int k = 0; k < 100; ++k) a += w2[t * 100 + k] * z1[k];
    z2[t] = tanhf(a);
  }
  __syncthreads();
  if (t < 16) {
    float a = b3[t];
    for (int k = 0; k < 100; ++k) a += w3[t * 100 + k] * z2[k];
    emb[(size_t)b * 16 + t] = a;
  }
}

// ---------------------------------------------------------------------------
extern "C" void kernel_launch(void* const* d_in, const int* in_sizes, int n_in,
                              void* d_out, int out_size, void* d_ws, size_t ws_size,
                              hipStream_t stream) {
  const float* obs       = (const float*)d_in[0];
  const int*   program_id= (const int*)  d_in[1];
  const int*   shape_id  = (const int*)  d_in[2];
  const int*   shape_id0 = (const int*)  d_in[3];
  const int*   shape_id1 = (const int*)  d_in[4];
  const float* eps_rp    = (const float*)d_in[5];
  const float* eps_rp0   = (const float*)d_in[6];
  const float* eps_rp1   = (const float*)d_in[7];
  const float* conv1_w   = (const float*)d_in[8];
  const float* conv1_b   = (const float*)d_in[9];
  const float* conv2_w   = (const float*)d_in[10];
  const float* conv2_b   = (const float*)d_in[11];
  const float* mlp_w1    = (const float*)d_in[12];
  const float* mlp_b1    = (const float*)d_in[13];
  const float* mlp_w2    = (const float*)d_in[14];
  const float* mlp_b2    = (const float*)d_in[15];
  const float* mlp_w3    = (const float*)d_in[16];
  const float* mlp_b3    = (const float*)d_in[17];
  const float* W_ih      = (const float*)d_in[18];
  const float* b_ih      = (const float*)d_in[19];
  const float* W_hh      = (const float*)d_in[20];
  const float* b_hh      = (const float*)d_in[21];
  const float* addr_emb  = (const float*)d_in[22];
  const float* pid_emb   = (const float*)d_in[23];
  const float* sid_emb   = (const float*)d_in[24];
  const float* pid_ext_w = (const float*)d_in[25];
  const float* pid_ext_b = (const float*)d_in[26];
  const float* sid_ext_w = (const float*)d_in[27];
  const float* sid_ext_b = (const float*)d_in[28];
  const float* rp_ext_w  = (const float*)d_in[29];
  const float* rp_ext_b  = (const float*)d_in[30];
  float* out = (float*)d_out;

  char* p = (char*)d_ws;
  auto alloc = [&](size_t bytes) -> void* {
    void* r = (void*)p;
    p += (bytes + 255) & ~(size_t)255;
    return r;
  };
  f16*   Wstep  = (f16*)  alloc((size_t)KSTEP * NGATE * 2);
  f16*   Wobs   = (f16*)  alloc((size_t)KOBS * NGATE * 2);
  float* addrc  = (float*)alloc(7 * 1024 * 4);
  f16*   obs16  = (f16*)  alloc((size_t)BATCH * KOBS * 2);
  float* gbase  = (float*)alloc((size_t)BATCH * NGATE * 4);
  float* gates  = (float*)alloc((size_t)BATCH * NGATE * 4);
  f16*   xin    = (f16*)  alloc((size_t)BATCH * KSTEP * 2);
  f16*   H[7];
  for (int s = 0; s < 7; ++s) H[s] = (f16*)alloc((size_t)BATCH * 256 * 2);
  float* C0     = (float*)alloc((size_t)BATCH * 256 * 4);
  float* CA     = (float*)alloc((size_t)BATCH * 256 * 4);
  float* CB     = (float*)alloc((size_t)BATCH * 256 * 4);
  float* rp_raw = (float*)alloc((size_t)BATCH * 4 * 4);
  float* rp0buf = (float*)alloc((size_t)BATCH * 2 * 4);
  float* rp0emb = (float*)alloc((size_t)BATCH * 16 * 4);

  // ---- one-time (per launch) weight prep ----
  k_pack_wstep<<<dim3(KSTEP / 32, 64), 256, 0, stream>>>(W_hh, W_ih, Wstep);
  k_pack_wobs <<<dim3(KOBS / 32, 64), 256, 0, stream>>>(W_ih, Wobs);
  k_addrconst<<<28, 256, 0, stream>>>(b_ih, b_hh, W_ih, addr_emb, addrc);

  // ---- conv tower -> obs_emb(f16, padded) ----
  k_conv<<<BATCH, 256, 0, stream>>>(obs, conv1_w, conv1_b, conv2_w, conv2_b, obs16);

  // ---- gate_base = obs_emb @ W_ih[:, :400]^T ----
  dim3 ggrid(BATCH / 16, NGATE / 128);
  k_gemm_wmma<KOBS / 32><<<ggrid, 256, 0, stream>>>(obs16, KOBS, Wobs, gbase);

  const int nbXin  = (BATCH * KSTEP + 255) / 256;
  const int nbElem = (BATCH * 256 + 255) / 256;
  auto step = [&](int s, const f16* hprev, int mode, const float* table,
                  const int* sel, const float* direct, const float* cprev,
                  float* cout) {
    k_build_xin<<<nbXin, 256, 0, stream>>>(xin, hprev, mode, table, sel, direct);
    k_gemm_wmma<KSTEP / 32><<<ggrid, 256, 0, stream>>>(xin, KSTEP, Wstep, gates);
    k_lstm<<<nbElem, 256, 0, stream>>>(gates, gbase, addrc + s * 1024, cprev,
                                       cout, H[s]);
  };

  // s=0: h0 = step(zeros, aid 0)
  step(0, nullptr, 0, nullptr, nullptr, nullptr, nullptr, C0);
  k_head<<<(BATCH * 3 + 255) / 256, 256, 0, stream>>>(H[0], pid_ext_w, pid_ext_b,
                                                      3, out, 0, OUTC);
  // s=1: h1 = step(pid_emb[program_id], aid 1, h0, c0)
  step(1, H[0], 1, pid_emb, program_id, nullptr, C0, CA);
  k_head<<<(BATCH * 2 + 255) / 256, 256, 0, stream>>>(H[1], sid_ext_w, sid_ext_b,
                                                      2, out, 3, OUTC);
  // s=2: h2 = step(sid_emb[shape_id], aid 4, h1, c1)
  step(2, H[1], 1, sid_emb, shape_id, nullptr, CA, CB);
  k_head<<<(BATCH * 4 + 255) / 256, 256, 0, stream>>>(H[2], rp_ext_w, rp_ext_b,
                                                      4, rp_raw, 0, 4);
  k_rp_post<<<32, 256, 0, stream>>>(rp_raw, eps_rp, out, 5, nullptr);
  // s=3: h1b = step(pid_emb[program_id], aid 2, h0, c0)
  step(3, H[0], 1, pid_emb, program_id, nullptr, C0, CA);
  k_head<<<(BATCH * 2 + 255) / 256, 256, 0, stream>>>(H[3], sid_ext_w, sid_ext_b,
                                                      2, out, 7, OUTC);
  // s=4: h2b = step(sid_emb[shape_id_0], aid 3, h1b, c1b)
  step(4, H[3], 1, sid_emb, shape_id0, nullptr, CA, CB);
  k_head<<<(BATCH * 2 + 255) / 256, 256, 0, stream>>>(H[4], sid_ext_w, sid_ext_b,
                                                      2, out, 9, OUTC);
  // s=5: h3b = step(sid_emb[shape_id_1], aid 5, h2b, c2b)
  step(5, H[4], 1, sid_emb, shape_id1, nullptr, CB, CA);
  k_head<<<(BATCH * 4 + 255) / 256, 256, 0, stream>>>(H[5], rp_ext_w, rp_ext_b,
                                                      4, rp_raw, 0, 4);
  k_rp_post<<<32, 256, 0, stream>>>(rp_raw, eps_rp0, out, 11, rp0buf);
  // MLP: rp0 -> rp0_emb
  k_mlp<<<BATCH, 128, 0, stream>>>(rp0buf, mlp_w1, mlp_b1, mlp_w2, mlp_b2,
                                   mlp_w3, mlp_b3, rp0emb);
  // s=6: h4b = step(rp0_emb, aid 6, h3b, c3b)
  step(6, H[5], 2, nullptr, nullptr, rp0emb, CA, CB);
  k_head<<<(BATCH * 4 + 255) / 256, 256, 0, stream>>>(H[6], rp_ext_w, rp_ext_b,
                                                      4, rp_raw, 0, 4);
  k_rp_post<<<32, 256, 0, stream>>>(rp_raw, eps_rp1, out, 13, nullptr);
}
